// GCNZinc_283467842549
// MI455X (gfx1250) — compile-verified, matrix operand order checked by reading, not run on hardware
//
#include <hip/hip_runtime.h>

typedef __attribute__((ext_vector_type(16))) _Float16 v16h;
typedef __attribute__((ext_vector_type(8)))  float    v8f;

constexpr int NN  = 100000;   // nodes (= 6250 * 16, no M edge tiles)
constexpr int NE  = 320000;   // edges
constexpr int DIN = 10;
constexpr int DH  = 256;      // hidden (= 16 * 16)
constexpr int ELEMS = NN * DH;

// ---------------- embedding: h = x @ W_emb + b_emb (K=10, memory-bound) ----
__global__ void __launch_bounds__(256)
emb_kernel(const float* __restrict__ x, const float* __restrict__ W,
           const float* __restrict__ b, float* __restrict__ h,
           _Float16* __restrict__ hh) {
  int tid = blockIdx.x * 256 + threadIdx.x;
  if (tid >= ELEMS) return;
  int n = tid >> 8, c = tid & 255;
  float acc = b[c];
  const float* xr = x + n * DIN;
#pragma unroll
  for (int k = 0; k < DIN; ++k) acc += xr[k] * W[k * DH + c];
  h[tid]  = acc;
  hh[tid] = (_Float16)acc;
}

// ---------------- degree / normalization ----------------------------------
__global__ void __launch_bounds__(256) deg_init(float* __restrict__ deg) {
  int n = blockIdx.x * 256 + threadIdx.x;
  if (n < NN) deg[n] = 1.0f;                       // self-loop
}
__global__ void __launch_bounds__(256)
deg_edges(const int* __restrict__ ei, float* __restrict__ deg) {
  int e = blockIdx.x * 256 + threadIdx.x;
  if (e < NE) unsafeAtomicAdd(&deg[ei[NE + e]], 1.0f);   // dst = row 1
}
__global__ void __launch_bounds__(256) deg_rsqrt(float* __restrict__ deg) {
  int n = blockIdx.x * 256 + threadIdx.x;
  if (n < NN) deg[n] = rsqrtf(deg[n]);
}

// ---------------- W (K x N, f32) -> Wt (N x K, f16) -----------------------
__global__ void __launch_bounds__(256)
wconv_kernel(const float* __restrict__ W, _Float16* __restrict__ Wt) {
  int tid = blockIdx.x * 256 + threadIdx.x;
  if (tid >= DH * DH) return;
  int n = tid >> 8, k = tid & 255;
  Wt[tid] = (_Float16)W[k * DH + n];
}

// ---------------- WMMA GEMM: C[M x 256] = A[M x 256] * Wt^T ---------------
// A: M x 256 f16 row-major.  Bt: 256 x 256 f16 row-major (= W transposed),
// so the B fragment load pattern matches the A fragment pattern.
// Each wave computes a 16 x 64 output strip: 1 A frag reused over 4 B frags,
// 8 k-steps of v_wmma_f32_16x16x32_f16, f32 accumulate.
__global__ void __launch_bounds__(256)
gemm_wmma_f16(const _Float16* __restrict__ A, const _Float16* __restrict__ Bt,
              float* __restrict__ C) {
  const int wid  = (blockIdx.x * 256 + threadIdx.x) >> 5;   // global wave id
  const int lane = threadIdx.x & 31;
  const int mt = wid >> 2;          // 0 .. 6249
  const int nt = wid & 3;           // 64-col block
  const int row0 = mt << 4;
  const int col0 = nt << 6;
  const int r    = lane & 15;       // row (A) / col (B) within tile
  const int hsel = lane >> 4;       // K sub-offset select (0 -> k, 1 -> k+8)

  const _Float16* arow = A + (size_t)(row0 + r) * DH + hsel * 8;

  v8f acc[4] = {v8f{}, v8f{}, v8f{}, v8f{}};

  for (int k0 = 0; k0 < 8; ++k0) {
    const int kb = k0 * 32;
    union { v16h h; int4 q[2]; } a;
    a.q[0] = *(const int4*)(arow + kb);        // K = kb+hsel*8 .. +7
    a.q[1] = *(const int4*)(arow + kb + 16);   // K = kb+hsel*8+16 .. +23
#pragma unroll
    for (int s = 0; s < 4; ++s) {
      const _Float16* brow = Bt + (size_t)(col0 + s * 16 + r) * DH + hsel * 8;
      union { v16h h; int4 q[2]; } bf;
      bf.q[0] = *(const int4*)(brow + kb);
      bf.q[1] = *(const int4*)(brow + kb + 16);
      acc[s] = __builtin_amdgcn_wmma_f32_16x16x32_f16(
          /*neg_a=*/false, a.h, /*neg_b=*/false, bf.h,
          /*c_mod=*/(short)0, acc[s], /*reuse_a=*/false, /*reuse_b=*/false);
    }
  }

  // C/D layout: lanes 0-15 -> M = v, lanes 16-31 -> M = 8 + v; N = lane & 15
  const int mbase = row0 + hsel * 8;
#pragma unroll
  for (int s = 0; s < 4; ++s) {
    const int cb = col0 + s * 16 + r;
#pragma unroll
    for (int v = 0; v < 8; ++v)
      C[(size_t)(mbase + v) * DH + cb] = acc[s][v];
  }
}

// ---------------- agg = hw * dinv^2  (self-loop term, also zero-fills) ----
__global__ void __launch_bounds__(256)
agg_init_kernel(const float* __restrict__ hw, const float* __restrict__ dinv,
                float* __restrict__ agg) {
  int tid = blockIdx.x * 256 + threadIdx.x;
  if (tid >= ELEMS) return;
  float di = dinv[tid >> 8];
  agg[tid] = hw[tid] * di * di;
}

// ---------------- edge scatter: agg[dst] += hw[src] * dinv[s]*dinv[d] -----
// 64 threads per edge, float4 per thread, 4 hardware f32 atomics.
__global__ void __launch_bounds__(256)
scatter_kernel(const int* __restrict__ ei, const float* __restrict__ hw,
               const float* __restrict__ dinv, float* __restrict__ agg) {
  int tid = blockIdx.x * 256 + threadIdx.x;
  int e = tid >> 6;
  if (e >= NE) return;
  int c = (tid & 63) << 2;
  int s = ei[e];
  int d = ei[NE + e];
  float norm = dinv[s] * dinv[d];
  float4 v = *(const float4*)(hw + (size_t)s * DH + c);
  float* p = agg + (size_t)d * DH + c;
  unsafeAtomicAdd(p + 0, v.x * norm);
  unsafeAtomicAdd(p + 1, v.y * norm);
  unsafeAtomicAdd(p + 2, v.z * norm);
  unsafeAtomicAdd(p + 3, v.w * norm);
}

// ---------------- epilogue: hout = relu(agg + b) + h  (+ f16 copy) --------
__global__ void __launch_bounds__(256)
epilogue_kernel(const float* __restrict__ agg, const float* __restrict__ b,
                const float* __restrict__ hin, float* __restrict__ hout,
                _Float16* __restrict__ hh) {
  int tid = blockIdx.x * 256 + threadIdx.x;
  if (tid >= ELEMS) return;
  float v = agg[tid] + b[tid & 255];
  v = fmaxf(v, 0.0f) + hin[tid];
  hout[tid] = v;
  if (hh) hh[tid] = (_Float16)v;
}

// --------------------------------------------------------------------------
extern "C" void kernel_launch(void* const* d_in, const int* in_sizes, int n_in,
                              void* d_out, int out_size, void* d_ws, size_t ws_size,
                              hipStream_t stream) {
  const float* x    = (const float*)d_in[0];
  const int*   ei   = (const int*)d_in[1];          // [2, NE], int
  const float* Wemb = (const float*)d_in[2];
  const float* bemb = (const float*)d_in[3];
  const float* Wl[3] = {(const float*)d_in[4], (const float*)d_in[6], (const float*)d_in[8]};
  const float* bl[3] = {(const float*)d_in[5], (const float*)d_in[7], (const float*)d_in[9]};

  char* ws = (char*)d_ws;
  size_t off = 0;
  float*    h    = (float*)(ws + off);    off += (size_t)ELEMS * 4;
  float*    hw   = (float*)(ws + off);    off += (size_t)ELEMS * 4;
  float*    agg  = (float*)(ws + off);    off += (size_t)ELEMS * 4;
  _Float16* hA   = (_Float16*)(ws + off); off += (size_t)ELEMS * 2;
  _Float16* Wt   = (_Float16*)(ws + off); off += (size_t)DH * DH * 2;
  float*    dinv = (float*)(ws + off);    off += (size_t)NN * 4;
  (void)ws_size; (void)n_in; (void)in_sizes; (void)out_size;

  const dim3 blk(256);
  const int gElem = (ELEMS + 255) / 256;       // 100000 blocks
  const int gNode = (NN + 255) / 256;
  const int gEdge = (NE + 255) / 256;
  const int gGemm = (NN / 16) * 4 / 8;         // 25000 waves / 8 per block = 3125
  const int gScat = (NE * 64) / 256;           // 80000

  // normalization (degree) — independent of layers, computed once
  deg_init<<<gNode, blk, 0, stream>>>(dinv);
  deg_edges<<<gEdge, blk, 0, stream>>>(ei, dinv);
  deg_rsqrt<<<gNode, blk, 0, stream>>>(dinv);

  // embedding
  emb_kernel<<<gElem, blk, 0, stream>>>(x, Wemb, bemb, h, hA);

  for (int l = 0; l < 3; ++l) {
    wconv_kernel<<<(DH * DH) / 256, blk, 0, stream>>>(Wl[l], Wt);
    gemm_wmma_f16<<<gGemm, blk, 0, stream>>>(hA, Wt, hw);
    agg_init_kernel<<<gElem, blk, 0, stream>>>(hw, dinv, agg);
    scatter_kernel<<<gScat, blk, 0, stream>>>(ei, hw, dinv, agg);
    const bool last = (l == 2);
    epilogue_kernel<<<gElem, blk, 0, stream>>>(
        agg, bl[l], h,
        last ? (float*)d_out : h,
        last ? (_Float16*)nullptr : hA);
  }
}
